// Transformer_73761768341667
// MI455X (gfx1250) — matrix operand order repaired and checked by reading, not compiled, not run on hardware
//
#include <hip/hip_runtime.h>
#include <cstdint>
#include <cstddef>

#define DEV __device__ __forceinline__

typedef __attribute__((ext_vector_type(16))) __bf16 v16bf;
typedef __attribute__((ext_vector_type(8)))  float  v8f;
typedef int v4i_vs __attribute__((vector_size(16)));   // matches builtin proto

static constexpr int B = 8, T = 2048, C = 768, H = 768;
static constexpr int BT = B * T;

// GEMM macro tile: 128x128, K-stage 64, 8 waves (256 thr), wave tile 32x64.
static constexpr int BM = 128, BN = 128, BK = 64;
static constexpr int LSTR = 72;   // padded LDS row stride in bf16 (144B = 16*9)

#if __has_builtin(__builtin_amdgcn_global_load_async_to_lds_b128)
#define USE_ASYNC 1
#else
#define USE_ASYNC 0
#endif

// ---------------------------------------------------------------- helpers

DEV unsigned short f32_bf16(float f) {
    unsigned int u = __float_as_uint(f);
    u += 0x7FFFu + ((u >> 16) & 1u);      // round-to-nearest-even
    return (unsigned short)(u >> 16);
}

union V16 { v16bf v; unsigned int u[8]; };

DEV v16bf pack2(uint4 q0, uint4 q1) {
    V16 r;
    r.u[0] = q0.x; r.u[1] = q0.y; r.u[2] = q0.z; r.u[3] = q0.w;
    r.u[4] = q1.x; r.u[5] = q1.y; r.u[6] = q1.z; r.u[7] = q1.w;
    return r.v;
}

// A-operand (16x32 bf16): lane half h holds K=[h*8,h*8+8) in VGPR0..3 and
// K=[16+h*8,16+h*8+8) in VGPR4..7 -> two ds_load_b128.
DEV v16bf lds_a(const unsigned short* buf, int row, int kk, int half) {
    const unsigned short* rp = buf + row * LSTR + kk;
    return pack2(*(const uint4*)(rp + half * 8),
                 *(const uint4*)(rp + 16 + half * 8));
}

// B-operand (32x16 bf16): lane half h holds K=[h*8,h*8+8) in VGPR0..3 and
// K=[16+h*8,16+h*8+8) in VGPR4..7 (same as A).
DEV v16bf lds_b(const unsigned short* buf, int row, int kk, int half) {
    const unsigned short* rp = buf + row * LSTR + kk;
    return pack2(*(const uint4*)(rp + half * 8),
                 *(const uint4*)(rp + 16 + half * 8));
}

// Stage a 128xBK bf16 tile global -> LDS (16KB = 1024 b128, 4 per thread).
DEV void stage(const unsigned short* __restrict__ G, int ldg, int row0, int k0,
               unsigned short* lbuf, int tid)
{
#pragma unroll
    for (int it = 0; it < 4; ++it) {
        int i   = tid + it * 256;
        int row = i >> 3;               // 64 halfs per row = 8 x b128
        int ko  = (i & 7) * 8;
        const unsigned short* gp = G + (size_t)(row0 + row) * ldg + (k0 + ko);
        unsigned short* lp = lbuf + row * LSTR + ko;
#if USE_ASYNC
        __builtin_amdgcn_global_load_async_to_lds_b128(
            (__attribute__((address_space(1))) v4i_vs*)gp,
            (__attribute__((address_space(3))) v4i_vs*)lp, 0, 0);
#else
        *(uint4*)lp = *(const uint4*)gp;
#endif
    }
}

DEV void wait_stage() {
#if USE_ASYNC
#if __has_builtin(__builtin_amdgcn_s_wait_asynccnt)
    __builtin_amdgcn_s_wait_asynccnt(0);
#else
    asm volatile("s_wait_asynccnt 0x0" ::: "memory");
#endif
#endif
    __syncthreads();
}

// ---------------------------------------------------------------- GEMM

enum { EP_BIAS_H = 0, EP_F32 = 1, EP_RES_DUAL = 2, EP_H = 3, EP_RES_RELU = 4 };

// Out[m,n] = epilogue( sum_k A[m,k] * Bm[n,k] ), double-buffered through LDS.
template<int EP>
__global__ __launch_bounds__(256)
void wmma_gemm(const unsigned short* __restrict__ A, int lda,
               const unsigned short* __restrict__ Bm, int ldb, int K,
               const float* __restrict__ bias,
               const float* __restrict__ res, int ldr,
               float* __restrict__ outF,
               unsigned short* __restrict__ outH, int ldo)
{
    __shared__ unsigned short As[2][BM * LSTR];
    __shared__ unsigned short Bs[2][BN * LSTR];

    const int tid  = threadIdx.x;
    const int lane = tid & 31;
    const int wave = tid >> 5;
    const int wm   = wave & 3;          // wave row   (4 x 32 rows)
    const int wn   = wave >> 2;         // wave col   (2 x 64 cols)
    const int lm   = lane & 15;
    const int half = lane >> 4;

    const int mblk = blockIdx.y * BM;
    const int nblk = blockIdx.x * BN;

    v8f acc[2][4] = {};

    stage(A,  lda, mblk, 0, As[0], tid);
    stage(Bm, ldb, nblk, 0, Bs[0], tid);
    wait_stage();

    const int KT = K / BK;
    for (int kt = 0; kt < KT; ++kt) {
        const int cur = kt & 1;
        if (kt + 1 < KT) {              // prefetch next stage into other buffer
            stage(A,  lda, mblk, (kt + 1) * BK, As[cur ^ 1], tid);
            stage(Bm, ldb, nblk, (kt + 1) * BK, Bs[cur ^ 1], tid);
        }
        const unsigned short* Ab = As[cur];
        const unsigned short* Bb = Bs[cur];
#pragma unroll
        for (int kk = 0; kk < BK; kk += 32) {
            v16bf a0 = lds_a(Ab, wm * 32 + lm,      kk, half);
            v16bf a1 = lds_a(Ab, wm * 32 + 16 + lm, kk, half);
#pragma unroll
            for (int tn = 0; tn < 4; ++tn) {
                v16bf b = lds_b(Bb, wn * 64 + tn * 16 + lm, kk, half);
                acc[0][tn] = __builtin_amdgcn_wmma_f32_16x16x32_bf16(
                    false, a0, false, b, (short)0, acc[0][tn], false, false);
                acc[1][tn] = __builtin_amdgcn_wmma_f32_16x16x32_bf16(
                    false, a1, false, b, (short)0, acc[1][tn], false, false);
            }
        }
        wait_stage();                   // next buffers landed + all waves done
    }

    // C/D layout: VGPR r, lanes 0-15 -> M=r, lanes 16-31 -> M=r+8.
#pragma unroll
    for (int sm = 0; sm < 2; ++sm) {
        const int rowb = mblk + wm * 32 + sm * 16 + 8 * half;
#pragma unroll
        for (int tn = 0; tn < 4; ++tn) {
            const int col = nblk + wn * 64 + tn * 16 + lm;
#pragma unroll
            for (int r = 0; r < 8; ++r) {
                const int    row = rowb + r;
                const size_t o   = (size_t)row * ldo + col;
                float v = acc[sm][tn][r];
                if (EP == EP_BIAS_H) {
                    outH[o] = f32_bf16(v + bias[col]);
                } else if (EP == EP_F32) {
                    outF[o] = v;
                } else if (EP == EP_RES_DUAL) {
                    float p = res[(size_t)row * ldr + col] + v;
                    outF[o] = p;
                    outH[o] = f32_bf16(p);
                } else if (EP == EP_H) {
                    outH[o] = f32_bf16(v);
                } else { // EP_RES_RELU
                    outF[o] = res[(size_t)row * ldr + col] + (v > 0.f ? v : 0.f);
                }
            }
        }
    }
}

// ---------------------------------------------------------------- small kernels

// x [B,T,C] f32 -> xb [B,T,C] bf16 and xbT [B,C,T] bf16 (LDS tile transpose)
__global__ __launch_bounds__(256)
void cvt_x(const float* __restrict__ x,
           unsigned short* __restrict__ xb,
           unsigned short* __restrict__ xbT)
{
    __shared__ unsigned short tile[32][33];
    const int b  = blockIdx.z;
    const int t0 = blockIdx.x * 32;
    const int c0 = blockIdx.y * 32;
    const int tx = threadIdx.x;          // 0..31
    const int ty = threadIdx.y;          // 0..7
#pragma unroll
    for (int j = 0; j < 4; ++j) {
        int t = t0 + ty + j * 8;
        size_t gi = ((size_t)b * T + t) * C + (c0 + tx);
        unsigned short h = f32_bf16(x[gi]);
        xb[gi] = h;
        tile[ty + j * 8][tx] = h;
    }
    __syncthreads();
#pragma unroll
    for (int j = 0; j < 4; ++j) {
        int c = c0 + ty + j * 8;
        xbT[((size_t)b * C + c) * T + (t0 + tx)] = tile[tx][ty + j * 8];
    }
}

__global__ void cvt(const float* __restrict__ in, unsigned short* __restrict__ o, int n)
{
    int idx = blockIdx.x * 256 + threadIdx.x;
    if (idx < n) o[idx] = f32_bf16(in[idx]);
}

// Column softmax stats (softmax over axis=1, i.e. over rows i for each col j),
// split into SL row-slices for parallelism, then combined.
static constexpr int SL = 8;

__global__ void colstats_part(const float* __restrict__ s, float2* __restrict__ part)
{
    int j  = blockIdx.x * 256 + threadIdx.x;   // column, coalesced
    int sl = blockIdx.y;
    float m = -3.402823466e38f, acc = 0.f;
    const int i0 = sl * (T / SL);
    for (int i = i0; i < i0 + T / SL; ++i) {
        float v = s[(size_t)i * T + j];
        if (v > m) { acc *= __expf(m - v); m = v; }
        acc += __expf(v - m);
    }
    part[(size_t)sl * T + j] = make_float2(m, acc);
}

__global__ void colstats_comb(const float2* __restrict__ part,
                              float* __restrict__ cmax, float* __restrict__ csum)
{
    int j = blockIdx.x * 256 + threadIdx.x;
    float m = -3.402823466e38f, acc = 0.f;
#pragma unroll
    for (int sl = 0; sl < SL; ++sl) {
        float2 p = part[(size_t)sl * T + j];
        if (p.x > m) { acc *= __expf(m - p.x); m = p.x; }
        acc += p.y * __expf(p.x - m);
    }
    cmax[j] = m;
    csum[j] = acc;
}

__global__ void make_attn(const float* __restrict__ s,
                          const float* __restrict__ cmax,
                          const float* __restrict__ csum,
                          unsigned short* __restrict__ attn)
{
    int idx = blockIdx.x * 256 + threadIdx.x;  // < T*T
    int j = idx & (T - 1);
    float a = __expf(s[idx] - cmax[j]) / csum[j];
    attn[idx] = f32_bf16(a);
}

// ---------------------------------------------------------------- launch

extern "C" void kernel_launch(void* const* d_in, const int* in_sizes, int n_in,
                              void* d_out, int out_size, void* d_ws, size_t ws_size,
                              hipStream_t stream)
{
    (void)in_sizes; (void)n_in; (void)out_size; (void)ws_size;

    const float* x  = (const float*)d_in[0];
    const float* Wk = (const float*)d_in[1];
    const float* bk = (const float*)d_in[2];
    const float* Wq = (const float*)d_in[3];
    const float* bq = (const float*)d_in[4];
    const float* W1 = (const float*)d_in[5];
    const float* W2 = (const float*)d_in[6];
    float* out = (float*)d_out;

    char* ws  = (char*)d_ws;
    size_t off = 0;
    auto alloc = [&](size_t bytes) -> void* {
        void* p = ws + off;
        off = (off + bytes + 255) & ~(size_t)255;
        return p;
    };

    const size_t BTC = (size_t)BT * C;
    const size_t HC  = (size_t)H * C;
    const size_t TT  = (size_t)T * T;

    unsigned short* xb     = (unsigned short*)alloc(BTC * 2);
    unsigned short* xbT    = (unsigned short*)alloc(BTC * 2);
    unsigned short* Kb     = (unsigned short*)alloc((size_t)BT * H * 2);
    unsigned short* Qb     = (unsigned short*)alloc((size_t)BT * H * 2);
    unsigned short* Wkb    = (unsigned short*)alloc(HC * 2);
    unsigned short* Wqb    = (unsigned short*)alloc(HC * 2);
    unsigned short* W1b    = (unsigned short*)alloc(HC * 2);
    unsigned short* W2b    = (unsigned short*)alloc(HC * 2);
    unsigned short* pb     = (unsigned short*)alloc(BTC * 2);
    float*          pF     = (float*)         alloc(BTC * 4);
    unsigned short* h1     = (unsigned short*)alloc(BTC * 2);
    float*          scores = (float*)         alloc(TT * 4);   // reused per batch
    unsigned short* attn   = (unsigned short*)alloc(TT * 2);   // reused per batch
    float2*         part   = (float2*)        alloc((size_t)SL * T * 8);
    float*          cmax   = (float*)         alloc((size_t)T * 4);
    float*          csum   = (float*)         alloc((size_t)T * 4);

    // 1) conversions
    cvt_x<<<dim3(T / 32, C / 32, B), dim3(32, 8), 0, stream>>>(x, xb, xbT);
    cvt<<<(int)(HC / 256), 256, 0, stream>>>(Wk, Wkb, (int)HC);
    cvt<<<(int)(HC / 256), 256, 0, stream>>>(Wq, Wqb, (int)HC);
    cvt<<<(int)(HC / 256), 256, 0, stream>>>(W1, W1b, (int)HC);
    cvt<<<(int)(HC / 256), 256, 0, stream>>>(W2, W2b, (int)HC);

    // 2) k = x@Wk^T + bk ; q = x@Wq^T + bq   (M=BT, N=H, K=C)
    dim3 gProj(H / BN, BT / BM);
    wmma_gemm<EP_BIAS_H><<<gProj, 256, 0, stream>>>(xb, C, Wkb, C, C, bk,
                                                    nullptr, 0, nullptr, Kb, H);
    wmma_gemm<EP_BIAS_H><<<gProj, 256, 0, stream>>>(xb, C, Wqb, C, C, bq,
                                                    nullptr, 0, nullptr, Qb, H);

    // 3) per batch: scores = K@Q^T -> column softmax -> p = x + attn@x
    for (int b = 0; b < B; ++b) {
        const unsigned short* Kbb = Kb + (size_t)b * T * H;
        const unsigned short* Qbb = Qb + (size_t)b * T * H;
        dim3 gScore(T / BN, T / BM);
        wmma_gemm<EP_F32><<<gScore, 256, 0, stream>>>(Kbb, H, Qbb, H, H,
                                                      nullptr, nullptr, 0,
                                                      scores, nullptr, T);
        colstats_part<<<dim3(T / 256, SL), 256, 0, stream>>>(scores, part);
        colstats_comb<<<T / 256, 256, 0, stream>>>(part, cmax, csum);
        make_attn<<<(int)(TT / 256), 256, 0, stream>>>(scores, cmax, csum, attn);

        dim3 gP(C / BN, T / BM);   // M=T, N=C, K=T
        wmma_gemm<EP_RES_DUAL><<<gP, 256, 0, stream>>>(
            attn, T, xbT + (size_t)b * C * T, T, T,
            nullptr, x + (size_t)b * T * C, C,
            pF + (size_t)b * T * C, pb + (size_t)b * T * C, C);
    }

    // 4) h1 = p@W1^T ; out = p + relu(h1@W2^T)   (M=BT, N=C, K=C)
    dim3 gMlp(C / BN, BT / BM);
    wmma_gemm<EP_H><<<gMlp, 256, 0, stream>>>(pb, C, W1b, C, C,
                                              nullptr, nullptr, 0, nullptr, h1, C);
    wmma_gemm<EP_RES_RELU><<<gMlp, 256, 0, stream>>>(h1, C, W2b, C, C,
                                                     nullptr, pF, C, out, nullptr, C);
}
